// FNPGNNv5_55052890800668
// MI455X (gfx1250) — matrix operand order repaired and checked by reading, not compiled, not run on hardware
//
#include <hip/hip_runtime.h>
#include <hip/hip_bf16.h>
#include <math.h>

static constexpr int cN  = 116;   // nodes
static constexpr int cB  = 1024;  // batch
static constexpr int cD  = 96;    // hidden
static constexpr int cH  = 4;     // heads
static constexpr int cDH = 24;    // head dim
static constexpr int cK  = 9;     // networks
static constexpr int cIN = 23;    // input dim
static constexpr int cCD = 960;   // concat dim
static constexpr int cNP = 128;   // padded N for WMMA

typedef __attribute__((ext_vector_type(16))) __bf16 v16bf;
typedef __attribute__((ext_vector_type(8)))  float  v8f;

__device__ __forceinline__ float eluf(float x) { return x > 0.f ? x : __expf(x) - 1.f; }
static constexpr float BN_SCALE = 0.999995000037499f; // 1/sqrt(1+1e-5)

// ---------------------------------------------------------------------------
// Generic fp32->bf16 WMMA GEMM: C[M,Nc] = epi(A[M,Kd] @ B[Kd,Nc])
// EPI: 0=none, 1=+bias, 2=elu(+bias), 3=bn_eval(elu(+bias), g, be)
// grid.x = groups of 8 m-tiles (one per wave), grid.y = n-tile.
// B panel staged once per block in dynamic LDS, transposed to col-major bf16
// so every WMMA B-fragment is 2 contiguous ds_load_b128. Full tiles take a
// branch-free store path.
// ---------------------------------------------------------------------------
template<int EPI>
__global__ void gemm_wmma(const float* __restrict__ A, const float* __restrict__ Bm,
                          const float* __restrict__ bias, const float* __restrict__ g,
                          const float* __restrict__ be, float* __restrict__ C,
                          int M, int Kd, int Nc, int lda, int ldb, int ldc, int Kpad)
{
    extern __shared__ __bf16 bp[];          // [16][Kpad], col-major: bp[col*Kpad + k]
    const int lane = threadIdx.x & 31;
    const int wave = threadIdx.x >> 5;
    const int n0 = blockIdx.y * 16;

    // cooperative B-panel load (coalesced 16-wide over columns)
    for (int idx = threadIdx.x; idx < 16 * Kpad; idx += blockDim.x) {
        int k = idx >> 4, col = idx & 15;
        float v = (k < Kd && (n0 + col) < Nc) ? Bm[(size_t)k * ldb + n0 + col] : 0.f;
        bp[(size_t)col * Kpad + k] = (__bf16)v;
    }
    __syncthreads();

    const int ntm = (M + 15) >> 4;
    const int mt = blockIdx.x * 8 + wave;
    if (mt >= ntm) return;
    const int m0 = mt * 16;
    const int hi = (lane >= 16) ? 1 : 0;
    const int row_a = m0 + (lane & 15);
    const int col_b = n0 + (lane & 15);
    const __bf16* bcol = bp + (size_t)(lane & 15) * Kpad;
    const bool fastA = ((lda & 3) == 0) && ((Kd & 31) == 0) && (row_a < M);

    v8f acc = {};
    for (int k0 = 0; k0 < Kd; k0 += 32) {
        v16bf af, bfm;
        const int abase = k0 + (hi ? 8 : 0);
        if (fastA) {
            const float4* p0 = reinterpret_cast<const float4*>(A + (size_t)row_a * lda + abase);
            const float4* p1 = reinterpret_cast<const float4*>(A + (size_t)row_a * lda + abase + 16);
            float4 x0 = p0[0], x1 = p0[1], y0 = p1[0], y1 = p1[1];
            af[0] = (__bf16)x0.x; af[1] = (__bf16)x0.y; af[2] = (__bf16)x0.z; af[3] = (__bf16)x0.w;
            af[4] = (__bf16)x1.x; af[5] = (__bf16)x1.y; af[6] = (__bf16)x1.z; af[7] = (__bf16)x1.w;
            af[8] = (__bf16)y0.x; af[9] = (__bf16)y0.y; af[10] = (__bf16)y0.z; af[11] = (__bf16)y0.w;
            af[12] = (__bf16)y1.x; af[13] = (__bf16)y1.y; af[14] = (__bf16)y1.z; af[15] = (__bf16)y1.w;
        } else {
#pragma unroll
            for (int e = 0; e < 16; ++e) {
                int ka = abase + e + ((e >= 8) ? 8 : 0);
                float av = (row_a < M && ka < Kd) ? A[(size_t)row_a * lda + ka] : 0.f;
                af[e] = (__bf16)av;
            }
        }
        const int bbase = k0 + (hi ? 8 : 0);
#pragma unroll
        for (int e = 0; e < 16; ++e) bfm[e] = bcol[bbase + e + ((e >= 8) ? 8 : 0)];   // two 8-wide chunks -> ds_load_b128 x2
        acc = __builtin_amdgcn_wmma_f32_16x16x32_bf16(false, af, false, bfm,
                                                      (short)0, acc, false, false);
    }

    auto epi = [&](float v, int col) -> float {
        if (EPI == 1)      v = v + bias[col];
        else if (EPI == 2) v = eluf(v + bias[col]);
        else if (EPI == 3) { v = eluf(v + bias[col]); v = v * (g[col] * BN_SCALE) + be[col]; }
        return v;
    };

    if ((m0 + 16 <= M) && (n0 + 16 <= Nc)) {
        // full tile: branch-free strided stores
        float* Crow = C + (size_t)(m0 + (hi ? 8 : 0)) * ldc + col_b;
#pragma unroll
        for (int r = 0; r < 8; ++r)
            Crow[(size_t)r * ldc] = epi(acc[r], col_b);
    } else {
#pragma unroll
        for (int r = 0; r < 8; ++r) {
            int row = m0 + r + (hi ? 8 : 0);
            if (row < M && col_b < Nc)
                C[(size_t)row * ldc + col_b] = epi(acc[r], col_b);
        }
    }
}

template<int EPI>
static void run_gemm(const float* A, const float* Bm, const float* bias, const float* g,
                     const float* be, float* C, int M, int Kd, int Nc,
                     int lda, int ldb, int ldc, hipStream_t stream)
{
    int Kpad = (Kd + 31) & ~31;
    dim3 grid((((M + 15) / 16) + 7) / 8, (Nc + 15) / 16);
    size_t smem = (size_t)16 * Kpad * sizeof(__bf16);
    gemm_wmma<EPI><<<grid, dim3(256), smem, stream>>>(A, Bm, bias, g, be, C,
                                                      M, Kd, Nc, lda, ldb, ldc, Kpad);
}

// ---------------------------------------------------------------------------
// GAT attention: one block per (batch, head). Row-softmax into LDS (bf16),
// then attn[116x116] @ Wh_head[116x24] via WMMA from LDS. attn never hits HBM.
// Wh head slice stored TRANSPOSED in LDS; accumulators staged in a separate
// LDS buffer and stored coalesced by the whole block (no predicated stores
// in the WMMA loop).
// ---------------------------------------------------------------------------
__global__ void gat_attention(const float* __restrict__ adj, const float* __restrict__ Wh,
                              const float* __restrict__ s, const float* __restrict__ t,
                              float* __restrict__ agg)
{
    const int b  = blockIdx.x;
    const int hd = blockIdx.y;
    const int tid = threadIdx.x;            // blockDim = 128 (4 waves)

    __shared__ __bf16 attn_lds[cNP][cNP];   // 32 KB   A-matrix (row-major over K=j)
    __shared__ __bf16 bmatT[32][cNP];       // 8 KB    B transposed: [col d][k=j]
    __shared__ float  tc[cNP];
    __shared__ float  outb[cNP][32];        // 16 KB   result staging

    if (tid < cNP) tc[tid] = (tid < cN) ? t[((size_t)b * cN + tid) * cH + hd] : 0.f;
    for (int idx = tid; idx < 32 * cNP; idx += 128) {
        int d = idx / cNP, j = idx % cNP;
        float v = (j < cN && d < cDH) ? Wh[((size_t)b * cN + j) * cD + hd * cDH + d] : 0.f;
        bmatT[d][j] = (__bf16)v;
    }
    for (int idx = tid; idx < (cNP - cN) * cNP; idx += 128)
        attn_lds[cN + idx / cNP][idx % cNP] = (__bf16)0.f;
    __syncthreads();

    if (tid < cN) {
        const int i = tid;
        const float si = s[((size_t)b * cN + i) * cH + hd];
        const float* arow = adj + ((size_t)b * cN + i) * cN;
        float mx = -3.0e38f;
        for (int j = 0; j < cN; ++j) {
            if (arow[j] != 0.f) {
                float e = si + tc[j];
                e = (e > 0.f) ? e : 0.2f * e;   // leaky_relu 0.2
                mx = fmaxf(mx, e);
            }
        }
        float sum = 0.f;
        for (int j = 0; j < cN; ++j) {
            float w = 0.f;
            if (arow[j] != 0.f) {
                float e = si + tc[j];
                e = (e > 0.f) ? e : 0.2f * e;
                w = __expf(e - mx);
                sum += w;
            }
            attn_lds[i][j] = (__bf16)w;
        }
        for (int j = cN; j < cNP; ++j) attn_lds[i][j] = (__bf16)0.f;
        const float inv = 1.f / sum;
        for (int j = 0; j < cN; ++j)
            attn_lds[i][j] = (__bf16)((float)attn_lds[i][j] * inv);
    }
    __syncthreads();

    const int lane = tid & 31;
    const int wave = tid >> 5;
    const int hi = (lane >= 16) ? 1 : 0;
#pragma unroll
    for (int q = 0; q < 4; ++q) {
        const int tt = wave + 4 * q;        // 16 tiles: 8 M-tiles x 2 N-tiles
        const int m0 = (tt >> 1) * 16;
        const int n0 = (tt & 1) * 16;
        const int rowA = m0 + (lane & 15);
        const int colB = n0 + (lane & 15);
        v8f acc = {};
#pragma unroll
        for (int k0 = 0; k0 < cNP; k0 += 32) {
            v16bf af, bfm;
            const int abase = k0 + (hi ? 8 : 0);
            const int bbase = k0 + (hi ? 8 : 0);
#pragma unroll
            for (int e = 0; e < 16; ++e) {
                af[e] = attn_lds[rowA][abase + e + ((e >= 8) ? 8 : 0)];
                bfm[e] = bmatT[colB][bbase + e + ((e >= 8) ? 8 : 0)];
            }
            acc = __builtin_amdgcn_wmma_f32_16x16x32_bf16(false, af, false, bfm,
                                                          (short)0, acc, false, false);
        }
#pragma unroll
        for (int r = 0; r < 8; ++r)
            outb[m0 + r + (hi ? 8 : 0)][colB] = acc[r];   // unconditional LDS stage
    }
    __syncthreads();
    // coalesced bounds-free store of the 116x24 block
    for (int idx = tid; idx < cN * cDH; idx += 128) {
        int row = idx / cDH, col = idx - row * cDH;
        agg[((size_t)b * cN + row) * cD + hd * cDH + col] = outb[row][col];
    }
}

// ------------------------- small f32 VALU kernels --------------------------

__global__ void mean_nodes(const float* __restrict__ h, const float* __restrict__ base,
                           float* __restrict__ out)
{
    int idx = blockIdx.x * blockDim.x + threadIdx.x;
    if (idx >= cB * cD) return;
    int d = idx % cD; size_t b = idx / cD;
    float sum = 0.f;
    for (int n = 0; n < cN; ++n) sum += h[(b * cN + n) * cD + d];
    out[idx] = (base ? base[d] : 0.f) + sum * (1.f / cN);
}

__global__ void compute_st(const float* __restrict__ Wh, const float* __restrict__ a,
                           float* __restrict__ s, float* __restrict__ t)
{
    int idx = blockIdx.x * blockDim.x + threadIdx.x;
    if (idx >= cB * cN * cH) return;
    int hh = idx % cH; size_t bn = idx / cH;
    const float* w  = Wh + bn * cD + hh * cDH;
    const float* av = a + hh * 2 * cDH;
    float ss = 0.f, tt = 0.f;
    for (int d = 0; d < cDH; ++d) { ss += w[d] * av[d]; tt += w[d] * av[cDH + d]; }
    s[idx] = ss; t[idx] = tt;
}

__global__ void gat_epilogue(const float* __restrict__ agg, const float* __restrict__ Wh,
                             const float* __restrict__ hres, const float* __restrict__ g,
                             const float* __restrict__ be, float* __restrict__ out, int n)
{
    int idx = blockIdx.x * blockDim.x + threadIdx.x;
    if (idx >= n) return;
    int d = idx % cD;
    float v = agg[idx] * (g[d] * BN_SCALE) + be[d];
    v = eluf(v) + Wh[idx];
    if (hres) v += hres[idx];
    out[idx] = v;
}

__global__ void concat2(const float* __restrict__ a, const float* __restrict__ bb,
                        float* __restrict__ c)
{
    int idx = blockIdx.x * blockDim.x + threadIdx.x;
    if (idx >= cB * 2 * cD) return;
    int cc = idx % (2 * cD); size_t b = idx / (2 * cD);
    c[idx] = (cc < cD) ? a[b * cD + cc] : bb[b * cD + cc - cD];
}

__global__ void add_vn(float* __restrict__ h, const float* __restrict__ vn)
{
    int idx = blockIdx.x * blockDim.x + threadIdx.x;
    if (idx >= cB * cN * cD) return;
    int d = idx % cD; size_t b = idx / ((size_t)cN * cD);
    h[idx] += 0.1f * vn[b * cD + d];
}

__global__ void pool_kernel(const float* __restrict__ h, const float* __restrict__ pool,
                            float* __restrict__ pooled)
{
    int idx = blockIdx.x * blockDim.x + threadIdx.x;
    if (idx >= cB * cK * cD) return;
    int d = idx % cD; int k = (idx / cD) % cK; size_t b = idx / (cD * cK);
    float acc = 0.f;
    for (int n = 0; n < cN; ++n) {
        float w = pool[n * cK + k];
        if (w != 0.f) acc += h[(b * cN + n) * cD + d] * w;
    }
    pooled[idx] = acc;
}

__global__ void net_attn(const float* __restrict__ pooled, const float* __restrict__ W_na1,
                         const float* __restrict__ b_na1, const float* __restrict__ W_na2,
                         const float* __restrict__ b_na2, float* __restrict__ netw)
{
    int b = blockIdx.x; int tid = threadIdx.x;     // blockDim = 288 = 9*32
    __shared__ float z1[cK][32];
    __shared__ float z2[cK];
    int k = tid >> 5, o = tid & 31;
    {
        float acc = b_na1[o];
        const float* pp = pooled + ((size_t)b * cK + k) * cD;
        for (int d = 0; d < cD; ++d) acc += pp[d] * W_na1[d * 32 + o];
        z1[k][o] = tanhf(acc);
    }
    __syncthreads();
    if (tid < cK) {
        float acc = b_na2[0];
        for (int o2 = 0; o2 < 32; ++o2) acc += z1[tid][o2] * W_na2[o2];
        z2[tid] = acc;
    }
    __syncthreads();
    if (tid == 0) {
        float mx = -3e38f;
        for (int kk = 0; kk < cK; ++kk) mx = fmaxf(mx, z2[kk]);
        float sum = 0.f, w[cK];
        for (int kk = 0; kk < cK; ++kk) { w[kk] = __expf(z2[kk] - mx); sum += w[kk]; }
        for (int kk = 0; kk < cK; ++kk) netw[(size_t)b * cK + kk] = w[kk] / sum;
    }
}

__global__ void flat_layernorm(const float* __restrict__ pooled, const float* __restrict__ netw,
                               const float* __restrict__ vnf, const float* __restrict__ g_ln,
                               const float* __restrict__ be_ln, float* __restrict__ flat)
{
    int b = blockIdx.x; int tid = threadIdx.x;     // blockDim = 256
    __shared__ float buf[cCD];
    __shared__ float red[8];
    float lsum = 0.f;
    for (int i = tid; i < cCD; i += 256) {
        float v;
        if (i < cK * cD) {
            int k = i / cD, d = i - k * cD;
            v = pooled[((size_t)b * cK + k) * cD + d] * netw[(size_t)b * cK + k];
        } else {
            v = vnf[(size_t)b * cD + (i - cK * cD)];
        }
        buf[i] = v; lsum += v;
    }
    for (int o = 16; o > 0; o >>= 1) lsum += __shfl_down(lsum, o, 32);
    if ((tid & 31) == 0) red[tid >> 5] = lsum;
    __syncthreads();
    if (tid == 0) { float ss = 0.f; for (int w = 0; w < 8; ++w) ss += red[w]; red[0] = ss; }
    __syncthreads();
    float mu = red[0] / cCD;
    __syncthreads();
    float lvar = 0.f;
    for (int i = tid; i < cCD; i += 256) { float d = buf[i] - mu; lvar += d * d; }
    for (int o = 16; o > 0; o >>= 1) lvar += __shfl_down(lvar, o, 32);
    if ((tid & 31) == 0) red[tid >> 5] = lvar;
    __syncthreads();
    if (tid == 0) { float ss = 0.f; for (int w = 0; w < 8; ++w) ss += red[w]; red[0] = ss; }
    __syncthreads();
    float inv = rsqrtf(red[0] / cCD + 1e-5f);
    for (int i = tid; i < cCD; i += 256)
        flat[(size_t)b * cCD + i] = (buf[i] - mu) * inv * g_ln[i] + be_ln[i];
}

__global__ void proj_norm(const float* __restrict__ pr, float* __restrict__ out)
{
    int b = blockIdx.x; int tid = threadIdx.x;     // blockDim = 64
    __shared__ float red[2];
    float v = pr[(size_t)b * 64 + tid];
    float sq = v * v;
    for (int o = 16; o > 0; o >>= 1) sq += __shfl_down(sq, o, 32);
    if ((tid & 31) == 0) red[tid >> 5] = sq;
    __syncthreads();
    float nrm = fmaxf(sqrtf(red[0] + red[1]), 1e-12f);
    out[(size_t)cB * 2 + (size_t)b * 64 + tid] = v / nrm;
}

// ---------------------------------------------------------------------------

extern "C" void kernel_launch(void* const* d_in, const int* in_sizes, int n_in,
                              void* d_out, int out_size, void* d_ws, size_t ws_size,
                              hipStream_t stream)
{
    const float* x     = (const float*)d_in[0];
    const float* adj   = (const float*)d_in[1];
    const float* W_enc = (const float*)d_in[2];
    const float* b_enc = (const float*)d_in[3];
    const float* g_in  = (const float*)d_in[4];
    const float* be_in = (const float*)d_in[5];
    const float* vn_emb= (const float*)d_in[6];
    const float* W1    = (const float*)d_in[7];
    const float* a1    = (const float*)d_in[8];
    const float* g1    = (const float*)d_in[9];
    const float* be1   = (const float*)d_in[10];
    const float* W2    = (const float*)d_in[11];
    const float* a2    = (const float*)d_in[12];
    const float* g2    = (const float*)d_in[13];
    const float* be2   = (const float*)d_in[14];
    const float* W_vn  = (const float*)d_in[15];
    const float* b_vn  = (const float*)d_in[16];
    const float* pool  = (const float*)d_in[17];
    const float* W_na1 = (const float*)d_in[18];
    const float* b_na1 = (const float*)d_in[19];
    const float* W_na2 = (const float*)d_in[20];
    const float* b_na2 = (const float*)d_in[21];
    const float* g_ln  = (const float*)d_in[22];
    const float* be_ln = (const float*)d_in[23];
    const float* W_c1  = (const float*)d_in[24];
    const float* b_c1  = (const float*)d_in[25];
    const float* W_c2  = (const float*)d_in[26];
    const float* b_c2  = (const float*)d_in[27];
    const float* W_p1  = (const float*)d_in[28];
    const float* b_p1  = (const float*)d_in[29];
    const float* W_p2  = (const float*)d_in[30];
    const float* b_p2  = (const float*)d_in[31];
    float* out = (float*)d_out;
    (void)in_sizes; (void)n_in; (void)out_size; (void)ws_size;

    // ---- workspace carve (~152 MB of fp32) ----
    float* ws = (float*)d_ws;
    size_t off = 0;
    auto carve = [&](size_t n) { float* p = ws + off; off += n; return p; };
    const size_t BN = (size_t)cB * cN;
    float* h    = carve(BN * cD);
    float* Wh   = carve(BN * cD);
    float* agg  = carve(BN * cD);
    float* sbuf = carve(BN * cH);
    float* tbuf = carve(BN * cH);
    float* vn0  = carve((size_t)cB * cD);
    float* vnN  = carve((size_t)cB * cD);
    float* vnF  = carve((size_t)cB * cD);
    float* mtmp = carve((size_t)cB * cD);
    float* cat  = carve((size_t)cB * 2 * cD);
    float* pooled = carve((size_t)cB * cK * cD);
    float* netw = carve((size_t)cB * cK);
    float* flat = carve((size_t)cB * cCD);
    float* c1   = carve((size_t)cB * 192);
    float* p1   = carve((size_t)cB * 128);
    float* pr   = carve((size_t)cB * 64);

    const int Mbn = (int)BN;

    // encoder: h = bn(elu(x @ W_enc + b_enc))
    run_gemm<3>(x, W_enc, b_enc, g_in, be_in, h, Mbn, cIN, cD, cIN, cD, cD, stream);
    // vn0 = vn_emb + mean_n(h)
    mean_nodes<<<(cB * cD + 255) / 256, 256, 0, stream>>>(h, vn_emb, vn0);

    // ---- GAT layer 1 ----
    run_gemm<0>(h, W1, nullptr, nullptr, nullptr, Wh, Mbn, cD, cD, cD, cD, cD, stream);
    compute_st<<<((int)(BN * cH) + 255) / 256, 256, 0, stream>>>(Wh, a1, sbuf, tbuf);
    gat_attention<<<dim3(cB, cH), 128, 0, stream>>>(adj, Wh, sbuf, tbuf, agg);
    gat_epilogue<<<((int)(BN * cD) + 255) / 256, 256, 0, stream>>>(agg, Wh, nullptr,
                                                                   g1, be1, h, (int)(BN * cD));
    // vn_new = elu([vn0, mean(h1)] @ W_vn + b_vn)
    mean_nodes<<<(cB * cD + 255) / 256, 256, 0, stream>>>(h, nullptr, mtmp);
    concat2<<<(cB * 2 * cD + 255) / 256, 256, 0, stream>>>(vn0, mtmp, cat);
    run_gemm<2>(cat, W_vn, b_vn, nullptr, nullptr, vnN, cB, 2 * cD, cD, 2 * cD, cD, cD, stream);
    add_vn<<<((int)(BN * cD) + 255) / 256, 256, 0, stream>>>(h, vnN);

    // ---- GAT layer 2 (+ residual h) ----
    run_gemm<0>(h, W2, nullptr, nullptr, nullptr, Wh, Mbn, cD, cD, cD, cD, cD, stream);
    compute_st<<<((int)(BN * cH) + 255) / 256, 256, 0, stream>>>(Wh, a2, sbuf, tbuf);
    gat_attention<<<dim3(cB, cH), 128, 0, stream>>>(adj, Wh, sbuf, tbuf, agg);
    gat_epilogue<<<((int)(BN * cD) + 255) / 256, 256, 0, stream>>>(agg, Wh, h,
                                                                   g2, be2, agg, (int)(BN * cD));
    float* h2 = agg;
    // vnF = elu([vnN, mean(h2)] @ W_vn + b_vn)
    mean_nodes<<<(cB * cD + 255) / 256, 256, 0, stream>>>(h2, nullptr, mtmp);
    concat2<<<(cB * 2 * cD + 255) / 256, 256, 0, stream>>>(vnN, mtmp, cat);
    run_gemm<2>(cat, W_vn, b_vn, nullptr, nullptr, vnF, cB, 2 * cD, cD, 2 * cD, cD, cD, stream);

    // pooling + network attention + layernorm
    pool_kernel<<<(cB * cK * cD + 255) / 256, 256, 0, stream>>>(h2, pool, pooled);
    net_attn<<<cB, 288, 0, stream>>>(pooled, W_na1, b_na1, W_na2, b_na2, netw);
    flat_layernorm<<<cB, 256, 0, stream>>>(pooled, netw, vnF, g_ln, be_ln, flat);

    // heads
    run_gemm<2>(flat, W_c1, b_c1, nullptr, nullptr, c1, cB, cCD, 192, cCD, 192, 192, stream);
    run_gemm<1>(c1, W_c2, b_c2, nullptr, nullptr, out, cB, 192, 2, 192, 2, 2, stream);
    run_gemm<2>(flat, W_p1, b_p1, nullptr, nullptr, p1, cB, cCD, 128, cCD, 128, 128, stream);
    run_gemm<1>(p1, W_p2, b_p2, nullptr, nullptr, pr, cB, 128, 64, 128, 64, 64, stream);
    proj_norm<<<cB, 64, 0, stream>>>(pr, out);
}
